// CombinedGNN_61263413510627
// MI455X (gfx1250) — matrix operand order repaired and checked by reading, not compiled, not run on hardware
//
#include <hip/hip_runtime.h>

// ---------------------------------------------------------------------------
// CombinedGNN on MI455X (gfx1250, wave32, WMMA).
// Precision: activations/weights in bf16 (halves the dominant edge-gather
// traffic vs fp32 and feeds v_wmma_f32_16x16x32_bf16 with fp32 accumulate).
// Aggregation scatter uses hardware fp32 global atomics.
// ---------------------------------------------------------------------------

#define NNODES 30000
#define NEDGES 960000
#define BN_EPS 1e-5f

typedef __attribute__((ext_vector_type(16))) __bf16 v16bf;
typedef __attribute__((ext_vector_type(8)))  float  v8f;

__device__ __forceinline__ unsigned short f2bf(float f) {
  unsigned int u = __float_as_uint(f);
  u += 0x7fffu + ((u >> 16) & 1u);       // round-to-nearest-even
  return (unsigned short)(u >> 16);
}
__device__ __forceinline__ float bf2f(unsigned short h) {
  return __uint_as_float(((unsigned int)h) << 16);
}
__device__ __forceinline__ void atomAddF(float* p, float v) {
  // lowers to global_atomic_add_f32 (no CAS loop)
  __hip_atomic_fetch_add(p, v, __ATOMIC_RELAXED, __HIP_MEMORY_SCOPE_AGENT);
}

// ---------------------------------------------------------------------------
// fp32 -> bf16 conversion
// ---------------------------------------------------------------------------
__global__ __launch_bounds__(256) void f32_to_bf16_k(const float* __restrict__ in,
                                                     unsigned short* __restrict__ out,
                                                     long long n) {
  long long stride = (long long)gridDim.x * blockDim.x;
  for (long long i = (long long)blockIdx.x * blockDim.x + threadIdx.x; i < n; i += stride)
    out[i] = f2bf(in[i]);
}

// ---------------------------------------------------------------------------
// degree count (dst)
// ---------------------------------------------------------------------------
__global__ __launch_bounds__(256) void degree_k(const int* __restrict__ dst,
                                                float* __restrict__ deg, int E) {
  int e = blockIdx.x * 256 + threadIdx.x;
  if (e < E) atomAddF(&deg[dst[e]], 1.0f);
}

// ---------------------------------------------------------------------------
// gather h[src] (bf16, 16B per thread) and scatter-add into agg[dst] (fp32)
// grid: (ceil(E/256), Din/8)
// ---------------------------------------------------------------------------
__global__ __launch_bounds__(256) void scatter_k(const unsigned short* __restrict__ h,
                                                 const int* __restrict__ src,
                                                 const int* __restrict__ dst,
                                                 float* __restrict__ agg,
                                                 int Din, int E) {
  int e = blockIdx.x * 256 + threadIdx.x;
  if (e >= E) return;
  int c0 = blockIdx.y * 8;
  int s = src[e], d = dst[e];
  uint4 v = *reinterpret_cast<const uint4*>(h + (size_t)s * Din + c0);
  const unsigned short* hv = reinterpret_cast<const unsigned short*>(&v);
  float* ap = agg + (size_t)d * Din + c0;
#pragma unroll
  for (int j = 0; j < 8; ++j) atomAddF(ap + j, bf2f(hv[j]));
}

// ---------------------------------------------------------------------------
// agg / clip(deg,1) -> bf16
// ---------------------------------------------------------------------------
__global__ __launch_bounds__(256) void mean_to_bf16_k(const float* __restrict__ agg,
                                                      const float* __restrict__ deg,
                                                      unsigned short* __restrict__ out,
                                                      int Din, long long n) {
  long long stride = (long long)gridDim.x * blockDim.x;
  for (long long i = (long long)blockIdx.x * blockDim.x + threadIdx.x; i < n; i += stride) {
    int row = (int)(i / Din);
    float dv = deg[row];
    dv = dv < 1.0f ? 1.0f : dv;
    out[i] = f2bf(agg[i] / dv);
  }
}

// ---------------------------------------------------------------------------
// WMMA bf16 GEMM: Out[M,Nc] = A0[M,K0]@W0[K0,Nc] (+ A1[M,K1]@W1[K1,Nc])
// Epilogue: if scale != null: Out = relu(y + bias[n]) * scale[n]; else Out = y.
// Tile 128x64, 256 threads = 8 waves (4x2), each wave 32x32 = 2x2 WMMA tiles.
// ---------------------------------------------------------------------------
#define BM 128
#define BN 64
#define BK 32
#define ASTR 40   // LDS row stride for A (bf16 elems), keeps 16B alignment
#define BSTR 72   // LDS row stride for B

__device__ __forceinline__ v16bf load_frag16(const unsigned short* p0,
                                             const unsigned short* p1) {
  union { uint4 u[2]; v16bf v; } t;
  t.u[0] = *reinterpret_cast<const uint4*>(p0);
  t.u[1] = *reinterpret_cast<const uint4*>(p1);
  return t.v;
}

__global__ __launch_bounds__(256) void gemm_bf16_k(
    const unsigned short* __restrict__ A0, const unsigned short* __restrict__ W0, int K0,
    const unsigned short* __restrict__ A1, const unsigned short* __restrict__ W1, int K1,
    float* __restrict__ Out, int Mrows, int Ncols,
    const float* __restrict__ bias, const float* __restrict__ scale) {
  __shared__ unsigned short As[BM * ASTR];
  __shared__ unsigned short Bs[BK * BSTR];

  const int tid   = threadIdx.x;
  const int lane  = tid & 31;
  const int wave  = tid >> 5;
  const int waveM = wave >> 1;  // 0..3
  const int waveN = wave & 1;   // 0..1
  const int blockM = blockIdx.x * BM;
  const int blockN = blockIdx.y * BN;

  v8f acc[2][2] = {};

  const int mloc  = lane & 15;
  const int kbase = (lane >> 4) << 3;   // 0 or 8 (A K-half)

  for (int part = 0; part < 2; ++part) {
    const unsigned short* A = part ? A1 : A0;
    const unsigned short* W = part ? W1 : W0;
    const int K = part ? K1 : K0;
    if (A == nullptr || K <= 0) continue;

    for (int k0 = 0; k0 < K; k0 += BK) {
      __syncthreads();  // previous tile fully consumed
      // --- load A tile (128x32 bf16): each thread 16 bf16 (32B) ---
      {
        int r  = tid >> 1;              // 0..127
        int hh = (tid & 1) << 4;        // 0 or 16
        int gr = blockM + r;
        uint4 a0 = {0, 0, 0, 0}, a1 = {0, 0, 0, 0};
        if (gr < Mrows) {
          const uint4* gp = reinterpret_cast<const uint4*>(A + (size_t)gr * K + (k0 + hh));
          a0 = gp[0];
          a1 = gp[1];
          if (k0 + BK < K)  // exercise global_prefetch_b8 on the next A tile
            __builtin_prefetch(A + (size_t)gr * K + (k0 + BK + hh), 0, 1);
        }
        uint4* sp = reinterpret_cast<uint4*>(&As[r * ASTR + hh]);
        sp[0] = a0;
        sp[1] = a1;
      }
      // --- load B tile (32x64 bf16): each thread 8 bf16 (16B) ---
      {
        int br   = tid >> 3;            // 0..31 (K row)
        int bseg = (tid & 7) << 3;      // 0..56 (N col)
        const uint4* gb = reinterpret_cast<const uint4*>(
            W + (size_t)(k0 + br) * Ncols + (blockN + bseg));
        *reinterpret_cast<uint4*>(&Bs[br * BSTR + bseg]) = *gb;
      }
      __syncthreads();

      // --- compute: one 16x16x32 WMMA per (tM,tN) tile ---
      v16bf afrag[2], bfrag[2];
#pragma unroll
      for (int tM = 0; tM < 2; ++tM) {
        int row = (waveM * 32 + tM * 16 + mloc) * ASTR;
        afrag[tM] = load_frag16(&As[row + kbase], &As[row + kbase + 16]);
      }
#pragma unroll
      for (int tN = 0; tN < 2; ++tN) {
        int col = waveN * 32 + tN * 16;
        union { unsigned short u[16]; v16bf v; } tb;
#pragma unroll
        for (int s = 0; s < 16; ++s)
          tb.u[s] = Bs[(kbase + (s & 7) + ((s >> 3) << 4)) * BSTR + col + mloc];
        bfrag[tN] = tb.v;
      }
#pragma unroll
      for (int tM = 0; tM < 2; ++tM)
#pragma unroll
        for (int tN = 0; tN < 2; ++tN)
          acc[tM][tN] = __builtin_amdgcn_wmma_f32_16x16x32_bf16(
              false, afrag[tM], false, bfrag[tN], (short)0, acc[tM][tN], false, false);
    }
  }

  // --- epilogue: C/D layout lane -> (N = lane%16, M = v + 8*(lane/16)) ---
#pragma unroll
  for (int tM = 0; tM < 2; ++tM) {
#pragma unroll
    for (int tN = 0; tN < 2; ++tN) {
      int n  = blockN + waveN * 32 + tN * 16 + (lane & 15);
      int mb = blockM + waveM * 32 + tM * 16 + ((lane >> 4) << 3);
      float bv = bias  ? bias[n]  : 0.0f;
      float sv = scale ? scale[n] : 0.0f;
#pragma unroll
      for (int v = 0; v < 8; ++v) {
        int m = mb + v;
        if (m < Mrows) {
          float y = acc[tM][tN][v];
          if (scale) y = fmaxf(y + bv, 0.0f) * sv;
          Out[(size_t)m * Ncols + n] = y;
        }
      }
    }
  }
}

// ---------------------------------------------------------------------------
// per-column sum / sumsq (for training-mode BatchNorm)
// grid: (ceil(Dout/256), rowChunks)
// ---------------------------------------------------------------------------
__global__ __launch_bounds__(256) void stats_k(const float* __restrict__ y,
                                               float* __restrict__ st,
                                               int Dout, int Nrows, int rowsPer) {
  int c = blockIdx.x * 256 + threadIdx.x;
  if (c >= Dout) return;
  int r0 = blockIdx.y * rowsPer;
  int r1 = r0 + rowsPer;
  if (r1 > Nrows) r1 = Nrows;
  float s = 0.0f, s2 = 0.0f;
  for (int r = r0; r < r1; ++r) {
    float v = y[(size_t)r * Dout + c];
    s += v;
    s2 += v * v;
  }
  atomAddF(&st[c], s);
  atomAddF(&st[Dout + c], s2);
}

// ---------------------------------------------------------------------------
// BN (biased var) + ReLU -> bf16
// ---------------------------------------------------------------------------
__global__ __launch_bounds__(256) void bn_relu_bf16_k(const float* __restrict__ y,
                                                      const float* __restrict__ st,
                                                      const float* __restrict__ gam,
                                                      const float* __restrict__ bet,
                                                      unsigned short* __restrict__ hout,
                                                      int Dout, float invN, long long n) {
  long long stride = (long long)gridDim.x * blockDim.x;
  for (long long i = (long long)blockIdx.x * blockDim.x + threadIdx.x; i < n; i += stride) {
    int c = (int)(i % Dout);
    float mean = st[c] * invN;
    float var  = st[Dout + c] * invN - mean * mean;
    float val  = gam[c] * (y[i] - mean) * rsqrtf(var + BN_EPS) + bet[c];
    hout[i] = f2bf(fmaxf(val, 0.0f));
  }
}

// ---------------------------------------------------------------------------
// scale[b][f] = softmax_b(attnW)[b][f] * W2[f]   (H = 1024, OUT = 1)
// ---------------------------------------------------------------------------
__global__ __launch_bounds__(256) void make_scale_k(const float* __restrict__ attnW,
                                                    const float* __restrict__ W2,
                                                    float* __restrict__ sbuf) {
  int f = blockIdx.x * 256 + threadIdx.x;
  if (f >= 1024) return;
  float a0 = attnW[f], a1 = attnW[1024 + f], a2 = attnW[2048 + f], a3 = attnW[3072 + f];
  float m  = fmaxf(fmaxf(a0, a1), fmaxf(a2, a3));
  float e0 = expf(a0 - m), e1 = expf(a1 - m), e2 = expf(a2 - m), e3 = expf(a3 - m);
  float inv = 1.0f / (e0 + e1 + e2 + e3);
  float w2 = W2[f];
  sbuf[f]        = e0 * inv * w2;
  sbuf[1024 + f] = e1 * inv * w2;
  sbuf[2048 + f] = e2 * inv * w2;
  sbuf[3072 + f] = e3 * inv * w2;
}

// ---------------------------------------------------------------------------
// out[row] += sum over 1024 columns of z[row,:]   (one block per row)
// ---------------------------------------------------------------------------
__global__ __launch_bounds__(256) void rowsum_acc_k(const float* __restrict__ z,
                                                    float* __restrict__ outAcc) {
  __shared__ float red[256];
  int row = blockIdx.x;
  const float* p = z + (size_t)row * 1024;
  float s = 0.0f;
  for (int c = threadIdx.x; c < 1024; c += 256) s += p[c];
  red[threadIdx.x] = s;
  __syncthreads();
  for (int off = 128; off > 0; off >>= 1) {
    if (threadIdx.x < off) red[threadIdx.x] += red[threadIdx.x + off];
    __syncthreads();
  }
  if (threadIdx.x == 0) outAcc[row] += red[0];
}

__global__ __launch_bounds__(256) void sigmoid_k(float* __restrict__ out,
                                                 const float* __restrict__ b2, int n) {
  int i = blockIdx.x * 256 + threadIdx.x;
  if (i < n) out[i] = 1.0f / (1.0f + expf(-(out[i] + b2[0])));
}

// ---------------------------------------------------------------------------
// host orchestration
// ---------------------------------------------------------------------------
extern "C" void kernel_launch(void* const* d_in, const int* in_sizes, int n_in,
                              void* d_out, int out_size, void* d_ws, size_t ws_size,
                              hipStream_t stream) {
  (void)in_sizes; (void)n_in; (void)ws_size;

  const float* x     = (const float*)d_in[0];
  const int*   ei    = (const int*)d_in[1];
  const float* Wl[4] = {(const float*)d_in[2], (const float*)d_in[7],
                        (const float*)d_in[12], (const float*)d_in[17]};
  const float* Wr[4] = {(const float*)d_in[3], (const float*)d_in[8],
                        (const float*)d_in[13], (const float*)d_in[18]};
  // bl{i} (d_in[4,9,14,19]) cancels inside BatchNorm -> skipped.
  const float* g[4]  = {(const float*)d_in[5], (const float*)d_in[10],
                        (const float*)d_in[15], (const float*)d_in[20]};
  const float* be[4] = {(const float*)d_in[6], (const float*)d_in[11],
                        (const float*)d_in[16], (const float*)d_in[21]};
  const float* Wfc   = (const float*)d_in[22];
  const float* bfc   = (const float*)d_in[23];
  const float* attnW = (const float*)d_in[24];
  const float* W2    = (const float*)d_in[25];
  const float* b2    = (const float*)d_in[26];
  float* out = (float*)d_out;

  const int dins[4]  = {1024, 1024, 512, 256};
  const int douts[4] = {1024, 512, 256, 128};

  // ---- carve workspace (256B aligned chunks) ----
  char* base = (char*)d_ws;
  auto carve = [&](size_t bytes) -> void* {
    void* p = (void*)base;
    base += (bytes + 255) & ~(size_t)255;
    return p;
  };
  unsigned short* xbf   = (unsigned short*)carve((size_t)NNODES * 1024 * 2);
  unsigned short* hbf   = (unsigned short*)carve((size_t)NNODES * 1024 * 2);
  unsigned short* aggbf = (unsigned short*)carve((size_t)NNODES * 1024 * 2);
  float* scr   = (float*)carve((size_t)NNODES * 1024 * 4);  // agg fp32 / GEMM out
  float* deg   = (float*)carve((size_t)NNODES * 4);
  float* stats = (float*)carve(2 * 1024 * 4);
  float* sbuf  = (float*)carve(4 * 1024 * 4);
  unsigned short* wWl[4];
  unsigned short* wWr[4];
  for (int l = 0; l < 4; ++l) {
    size_t nw = (size_t)4 * dins[l] * douts[l];
    wWl[l] = (unsigned short*)carve(nw * 2);
    wWr[l] = (unsigned short*)carve(nw * 2);
  }
  unsigned short* wFc = (unsigned short*)carve((size_t)4 * 128 * 1024 * 2);

  // ---- one-time conversions / precompute ----
  hipMemsetAsync(out, 0, (size_t)out_size * sizeof(float), stream);

  {
    long long nx = (long long)NNODES * 1024;
    f32_to_bf16_k<<<(int)((nx + 255) / 256), 256, 0, stream>>>(x, xbf, nx);
  }
  for (int l = 0; l < 4; ++l) {
    long long nw = (long long)4 * dins[l] * douts[l];
    f32_to_bf16_k<<<(int)((nw + 255) / 256), 256, 0, stream>>>(Wl[l], wWl[l], nw);
    f32_to_bf16_k<<<(int)((nw + 255) / 256), 256, 0, stream>>>(Wr[l], wWr[l], nw);
  }
  {
    long long nw = (long long)4 * 128 * 1024;
    f32_to_bf16_k<<<(int)((nw + 255) / 256), 256, 0, stream>>>(Wfc, wFc, nw);
  }
  make_scale_k<<<4, 256, 0, stream>>>(attnW, W2, sbuf);

  // ---- 4 branches ----
  for (int b = 0; b < 4; ++b) {
    const int* src = ei + (size_t)b * 2 * NEDGES;
    const int* dst = src + NEDGES;

    hipMemsetAsync(deg, 0, (size_t)NNODES * 4, stream);
    degree_k<<<(NEDGES + 255) / 256, 256, 0, stream>>>(dst, deg, NEDGES);

    const unsigned short* hcur = xbf;
    for (int l = 0; l < 4; ++l) {
      const int Din = dins[l], Dout = douts[l];

      // mean aggregation
      hipMemsetAsync(scr, 0, (size_t)NNODES * Din * 4, stream);
      dim3 sg((NEDGES + 255) / 256, Din / 8);
      scatter_k<<<sg, 256, 0, stream>>>(hcur, src, dst, scr, Din, NEDGES);
      long long na = (long long)NNODES * Din;
      mean_to_bf16_k<<<(int)((na + 255) / 256), 256, 0, stream>>>(scr, deg, aggbf, Din, na);

      // fused SAGE GEMM: y = agg@Wl + h@Wr
      dim3 gg((NNODES + BM - 1) / BM, Dout / BN);
      gemm_bf16_k<<<gg, 256, 0, stream>>>(
          aggbf, wWl[l] + (size_t)b * Din * Dout, Din,
          hcur,  wWr[l] + (size_t)b * Din * Dout, Din,
          scr, NNODES, Dout, nullptr, nullptr);

      // BatchNorm stats + apply + ReLU -> bf16
      hipMemsetAsync(stats, 0, (size_t)2 * Dout * 4, stream);
      dim3 stg((Dout + 255) / 256, 60);
      stats_k<<<stg, 256, 0, stream>>>(scr, stats, Dout, NNODES, 500);
      long long ny = (long long)NNODES * Dout;
      bn_relu_bf16_k<<<(int)((ny + 255) / 256), 256, 0, stream>>>(
          scr, stats, g[l] + (size_t)b * Dout, be[l] + (size_t)b * Dout,
          hbf, Dout, 1.0f / (float)NNODES, ny);
      hcur = hbf;
    }

    // FC GEMM with fused bias + ReLU + attention/W2 column scale
    dim3 gfc((NNODES + BM - 1) / BM, 1024 / BN);
    gemm_bf16_k<<<gfc, 256, 0, stream>>>(
        hcur, wFc + (size_t)b * 128 * 1024, 128,
        nullptr, nullptr, 0,
        scr, NNODES, 1024, bfc + (size_t)b * 1024, sbuf + (size_t)b * 1024);
    rowsum_acc_k<<<NNODES, 256, 0, stream>>>(scr, out);
  }

  sigmoid_k<<<(NNODES + 255) / 256, 256, 0, stream>>>(out, b2, NNODES);
}
